// GNNDi_53257594470738
// MI455X (gfx1250) — matrix operand order repaired and checked by reading, not compiled, not run on hardware
//
#include <hip/hip_runtime.h>
#include <stdint.h>

constexpr int NB = 2, NV = 256, NH = 128, NL = 3, NG = 32;
constexpr int NVV = NV * NV;
constexpr float EPSF = 1e-5f;

typedef __attribute__((ext_vector_type(16))) __bf16 v16bf;
typedef __attribute__((ext_vector_type(8)))  float  v8f;
typedef unsigned int u32x4 __attribute__((ext_vector_type(4)));
typedef int          i32x4 __attribute__((ext_vector_type(4)));
typedef int          i32x8 __attribute__((ext_vector_type(8)));

union Frag16 { v16bf v; uint4 q[2]; };

#if __has_builtin(__builtin_amdgcn_tensor_load_to_lds)
#define USE_TDM 1
#else
#define USE_TDM 0
#endif

__device__ __forceinline__ __bf16 f2bf(float f) {
  union { float f; uint32_t u; } c; c.f = f;
  uint32_t r = c.u + 0x7FFFu + ((c.u >> 16) & 1u);  // RNE
  union { uint16_t s; __bf16 b; } o; o.s = (uint16_t)(r >> 16);
  return o.b;
}

__device__ __forceinline__ float warp_sum32(float x) {
#pragma unroll
  for (int m = 16; m > 0; m >>= 1) x += __shfl_xor(x, m, 32);
  return x;
}

// ---------------------------------------------------------------------------
// Tile staging: global (rows x 128 bf16, row stride 128) -> LDS rows padded to
// 272B (136 bf16).  TDM path: one tensor_load_to_lds per tile, LDS padding via
// D#.pad (interval 64 DWORDs -> code 5, amount 4 DWORDs -> code 3).
// Fallback: per-lane global_load_async_to_lds_b128 (ASYNCcnt).
// ---------------------------------------------------------------------------
#if USE_TDM
__device__ __forceinline__ void tdm_load_tile(uint32_t lds_off,
                                              const void* gptr, uint32_t rows) {
  const uint64_t ga = (uint64_t)gptr;
  u32x4 g0;
  g0[0] = 1u;                                   // count=1 valid descriptor
  g0[1] = lds_off;                              // LDS byte address
  g0[2] = (uint32_t)ga;                         // global_addr[31:0]
  g0[3] = (uint32_t)((ga >> 32) & 0x1FFFFFFu) | (2u << 30);  // [56:32], type=2
  i32x8 g1;
  // data_size=2B (code 1), pad_enable, pad_interval=64 DW (5), pad_amount=4 DW (3)
  g1[0] = (int)((1u << 16) | (1u << 20) | (5u << 22) | (3u << 25));
  g1[1] = (int)(128u << 16);                    // tensor_dim0 = 128 @ bit48
  g1[2] = (int)((rows & 0xFFFFu) << 16);        // tensor_dim1 = rows @ bit80
  g1[3] = (int)((rows >> 16) | (128u << 16));   // tile_dim0 = 128 @ bit112
  g1[4] = (int)(rows & 0xFFFFu);                // tile_dim1 = rows, tile_dim2=0
  g1[5] = (int)128;                             // tensor_dim0_stride = 128
  g1[6] = 0;
  g1[7] = 0;
  i32x4 z4 = {0, 0, 0, 0};
#if __clang_major__ >= 23
  i32x8 z8 = {0, 0, 0, 0, 0, 0, 0, 0};
  __builtin_amdgcn_tensor_load_to_lds(g0, g1, z4, z4, z8, 0);
#else
  __builtin_amdgcn_tensor_load_to_lds(g0, g1, z4, z4, 0);
#endif
}
#define STAGE_TILE(ldsbase, gsrc, rows, t)                     \
  do {                                                         \
    if ((t) == 0) tdm_load_tile((ldsbase), (gsrc), (rows));    \
  } while (0)
#define STAGE_WAIT(t)                                          \
  do {                                                         \
    if ((t) == 0) __builtin_amdgcn_s_wait_tensorcnt(0);        \
  } while (0)
#else
__device__ __forceinline__ void async_copy_tile(uint32_t lds_base,
                                                const __bf16* gbase, int rows,
                                                int t) {
  const int chunks = rows * 8;  // 8 x 16B chunks (128B) copied per row
  for (int c = t; c < chunks; c += 128) {
    const int m = c >> 3, kc = c & 7;
    const uint32_t dst = lds_base + (uint32_t)m * 272u + (uint32_t)kc * 16u;
    const void* src = (const void*)(gbase + m * 128 + kc * 8);
    asm volatile("global_load_async_to_lds_b128 %0, %1, off" ::"v"(dst),
                 "v"(src)
                 : "memory");
  }
}
#define STAGE_TILE(ldsbase, gsrc, rows, t) \
  async_copy_tile((ldsbase), (const __bf16*)(gsrc), (rows), (t))
#define STAGE_WAIT(t) asm volatile("s_wait_asynccnt 0" ::: "memory")
#endif

// ---------------------------------------------------------------------------
// Shared WMMA core: C(64x128) = A(64x128) * B(128x128), bf16 in, f32 acc.
// sA: [m][k] (row stride 136), sBt: B transposed [n][k] (row stride 136).
// Block = 128 threads = 4 waves; wave w owns M rows [16w,16w+16).
// B fragments are software-pipelined (load nt+1 before issuing WMMA nt) so
// the LDS latency hides under the XDL pipe instead of s_wait_dscnt 0 stalls.
// ---------------------------------------------------------------------------
__device__ __forceinline__ void gemm_tile(__bf16 (*sA)[136], __bf16 (*sBt)[136],
                                          v8f* acc) {
  const int t = threadIdx.x;
  const int w = t >> 5, lane = t & 31;
  const int half = lane >> 4, lq = lane & 15;
  const __bf16* arow = &sA[16 * w + lq][0];
#pragma unroll
  for (int kt = 0; kt < 4; ++kt) {
    const int k0 = kt * 32;
    Frag16 a;
    a.q[0] = *(const uint4*)(arow + k0 + 8 * half);
    a.q[1] = *(const uint4*)(arow + k0 + 16 + 8 * half);
    Frag16 bcur;
    {
      const __bf16* p = &sBt[lq][0] + k0 + 8 * half;
      bcur.q[0] = *(const uint4*)(p);
      bcur.q[1] = *(const uint4*)(p + 16);
    }
#pragma unroll
    for (int nt = 0; nt < 8; ++nt) {
      Frag16 bnxt;
      if (nt < 7) {
        const __bf16* p = &sBt[(nt + 1) * 16 + lq][0] + k0 + 8 * half;
        bnxt.q[0] = *(const uint4*)(p);
        bnxt.q[1] = *(const uint4*)(p + 16);
      }
      acc[nt] = __builtin_amdgcn_wmma_f32_16x16x32_bf16(
          false, a.v, false, bcur.v, (short)0, acc[nt], false, false);
      if (nt < 7) bcur = bnxt;
    }
  }
}

// ---------------------------------------------------------------------------
// One-time weight prep: bf16 transposed [n][k] copies of W_ee, Wc[l], plo_W[l].
// ---------------------------------------------------------------------------
__global__ __launch_bounds__(256) void k_prep_w(
    const float* __restrict__ Wee, const float* __restrict__ Wc,
    const float* __restrict__ ploW, __bf16* __restrict__ wbf) {
  const int mat = blockIdx.x;  // 0: Wee, 1..3: Wc[l], 4..6: plo_W[l]
  const float* src = (mat == 0) ? Wee
                     : (mat <= 3) ? Wc + (size_t)(mat - 1) * NH * NH
                                  : ploW + (size_t)(mat - 4) * NH * NH;
  __bf16* dst = wbf + (size_t)mat * NH * NH;
  for (int idx = threadIdx.x; idx < NH * NH; idx += 256) {
    const int n = idx >> 7, k = idx & 127;
    dst[n * NH + k] = f2bf(src[k * NH + n]);
  }
}

// ---------------------------------------------------------------------------
// Node embedding: h[b,v,:] = emb_type[x[b,v]] @ W_node + b_node
// ---------------------------------------------------------------------------
__global__ __launch_bounds__(128) void k_node_embed(
    const int* __restrict__ x, const float* __restrict__ emb,
    const float* __restrict__ Wn, const float* __restrict__ bn,
    float* __restrict__ h) {
  __shared__ float se[NH];
  const int row = blockIdx.x, o = threadIdx.x;
  se[o] = emb[x[row] * NH + o];
  __syncthreads();
  float acc = bn[o];
  for (int k = 0; k < NH; ++k) acc += se[k] * Wn[k * NH + o];
  h[row * NH + o] = acc;
}

// ---------------------------------------------------------------------------
// Edge embedding: e = (adj^T(6ch) @ W_eattr + b_eattr) @ W_ee + b_ee
// ---------------------------------------------------------------------------
__global__ __launch_bounds__(128) void k_edge_embed(
    const float* __restrict__ adj, const float* __restrict__ We,
    const float* __restrict__ be, const __bf16* __restrict__ weet,
    const float* __restrict__ bee, float* __restrict__ e,
    __bf16* __restrict__ ebf) {
  __shared__ __align__(16) __bf16 sA[64][136];
  __shared__ __align__(16) __bf16 sBt[128][136];
  __shared__ float sWe[6 * NH], sbe[NH], sbee[NH], sAdj[64][6];
  const long row0 = (long)blockIdx.x * 64;
  const int b = (int)(row0 / NVV);
  const long rem0 = row0 % NVV;
  const int t = threadIdx.x;

  STAGE_TILE((uint32_t)(uintptr_t)&sBt[0][0], weet, 128, t);
  for (int idx = t; idx < 6 * NH; idx += 128) sWe[idx] = We[idx];
  if (t < NH) { sbe[t] = be[t]; sbee[t] = bee[t]; }
  for (int idx = t; idx < 64 * 6; idx += 128) {
    int m = idx / 6, c = idx % 6;
    sAdj[m][c] = adj[((long)b * 6 + c) * NVV + rem0 + m];
  }
  STAGE_WAIT(t);
  __syncthreads();
  for (int idx = t; idx < 64 * NH; idx += 128) {
    int m = idx >> 7, k = idx & 127;
    float a = sbe[k];
#pragma unroll
    for (int c = 0; c < 6; ++c) a += sAdj[m][c] * sWe[c * NH + k];
    sA[m][k] = f2bf(a);
  }
  __syncthreads();

  v8f acc[8];
#pragma unroll
  for (int nt = 0; nt < 8; ++nt)
#pragma unroll
    for (int r = 0; r < 8; ++r) acc[nt][r] = 0.f;
  gemm_tile(sA, sBt, acc);

  const int w = t >> 5, lane = t & 31;
  const int half = lane >> 4, lq = lane & 15;
#pragma unroll
  for (int nt = 0; nt < 8; ++nt) {
    const int n = nt * 16 + lq;
#pragma unroll
    for (int r = 0; r < 8; ++r) {
      const int ml = 16 * w + r + 8 * half;
      const long row = row0 + ml;
      const float val = acc[nt][r] + sbee[n];
      e[row * NH + n] = val;
      ebf[row * NH + n] = f2bf(val);
    }
  }
}

// ---------------------------------------------------------------------------
// Per-layer node GEMMs: Uh/Vh/Ah/Bh = h@W*+b*, and zero agg.
// ---------------------------------------------------------------------------
__global__ __launch_bounds__(128) void k_hmm(
    const float* __restrict__ h, const float* __restrict__ Wu,
    const float* __restrict__ bu, const float* __restrict__ Wv,
    const float* __restrict__ bv, const float* __restrict__ Wa,
    const float* __restrict__ ba, const float* __restrict__ Wb,
    const float* __restrict__ bbp, float* __restrict__ Uh,
    float* __restrict__ Vh, float* __restrict__ Ah, float* __restrict__ Bh,
    float* __restrict__ agg, int l) {
  __shared__ float sh[NH];
  const int row = blockIdx.x, o = threadIdx.x;
  sh[o] = h[row * NH + o];
  __syncthreads();
  const float* wu = Wu + (long)l * NH * NH;
  const float* wv = Wv + (long)l * NH * NH;
  const float* wa = Wa + (long)l * NH * NH;
  const float* wb = Wb + (long)l * NH * NH;
  float au = bu[l * NH + o], av = bv[l * NH + o];
  float aa = ba[l * NH + o], ab = bbp[l * NH + o];
  for (int k = 0; k < NH; ++k) {
    const float hk = sh[k];
    au += hk * wu[k * NH + o];
    av += hk * wv[k * NH + o];
    aa += hk * wa[k * NH + o];
    ab += hk * wb[k * NH + o];
  }
  Uh[row * NH + o] = au;
  Vh[row * NH + o] = av;
  Ah[row * NH + o] = aa;
  Bh[row * NH + o] = ab;
  agg[row * NH + o] = 0.f;
}

// ---------------------------------------------------------------------------
// Big fused kernel: e_new = e@Wc + bc + Ah[j] + Bh[i]; gates = sigmoid*mask;
// agg[b,i,:] += sum_j gates*Vh[j].  Block = fixed (b,i), 64-wide j tile.
// ---------------------------------------------------------------------------
__global__ __launch_bounds__(128) void k_edge_gemm_gate(
    const __bf16* __restrict__ ebf, const __bf16* __restrict__ wct,
    const float* __restrict__ bc, const float* __restrict__ Ah,
    const float* __restrict__ Bh, const float* __restrict__ Vhp,
    const float* __restrict__ adj, float* __restrict__ enew,
    float* __restrict__ agg, int l) {
  __shared__ __align__(16) __bf16 sA[64][136];
  __shared__ __align__(16) __bf16 sBt[128][136];
  __shared__ float sBh[NH], sbc[NH], smask[64], sagg[NH];

  const int t = threadIdx.x;
  const int blk = blockIdx.x;
  const int jt = blk & 3;
  const int bi = blk >> 2;
  const int b = bi >> 8, i = bi & 255;
  const int j0 = jt * 64;
  const long rowbase = ((long)(b * NV + i)) * NV + j0;

  STAGE_TILE((uint32_t)(uintptr_t)&sA[0][0], ebf + rowbase * NH, 64, t);
  STAGE_TILE((uint32_t)(uintptr_t)&sBt[0][0], wct, 128, t);
  if (t < NH) {
    sBh[t] = Bh[(b * NV + i) * NH + t];
    sbc[t] = bc[l * NH + t];
    sagg[t] = 0.f;
  }
  if (t < 64) smask[t] = adj[((long)b * 6) * NVV + i * NV + j0 + t];
  STAGE_WAIT(t);
  __syncthreads();

  v8f acc[8];
#pragma unroll
  for (int nt = 0; nt < 8; ++nt)
#pragma unroll
    for (int r = 0; r < 8; ++r) acc[nt][r] = 0.f;
  gemm_tile(sA, sBt, acc);

  const int w = t >> 5, lane = t & 31;
  const int half = lane >> 4, lq = lane & 15;
#pragma unroll
  for (int nt = 0; nt < 8; ++nt) {
    const int n = nt * 16 + lq;
    const float bhn = sBh[n], bcn = sbc[n];
    float aggp = 0.f;
#pragma unroll
    for (int r = 0; r < 8; ++r) {
      const int ml = 16 * w + r + 8 * half;          // local j
      const long jrowA = (long)b * NV + j0 + ml;     // row into Ah/Vh
      const float c = acc[nt][r] + bcn + bhn + Ah[jrowA * NH + n];
      enew[(rowbase + ml) * NH + n] = c;
      const float g = smask[ml] / (1.f + __expf(-c));
      aggp += g * Vhp[jrowA * NH + n];
    }
    atomicAdd(&sagg[n], aggp);  // ds_add_f32
  }
  __syncthreads();
  if (t < NH) atomicAdd(&agg[((long)b * NV + i) * NH + t], sagg[t]);
}

// ---------------------------------------------------------------------------
// h update: h = h_in + relu(LN(Uh + agg)).  One wave per row.
// ---------------------------------------------------------------------------
__global__ __launch_bounds__(256) void k_h_update(
    const float* __restrict__ Uh, const float* __restrict__ agg,
    const float* __restrict__ lnh_g, const float* __restrict__ lnh_b,
    float* __restrict__ h, int l) {
  const int warp = threadIdx.x >> 5, lane = threadIdx.x & 31;
  const long row = (long)blockIdx.x * 8 + warp;
  float x[4], hin[4];
#pragma unroll
  for (int u = 0; u < 4; ++u) {
    const int c = lane * 4 + u;
    x[u] = Uh[row * NH + c] + agg[row * NH + c];
    hin[u] = h[row * NH + c];
  }
  float s = warp_sum32(x[0] + x[1] + x[2] + x[3]);
  const float mean = s * (1.f / NH);
  float vs = 0.f;
#pragma unroll
  for (int u = 0; u < 4; ++u) { float d = x[u] - mean; vs += d * d; }
  vs = warp_sum32(vs);
  const float rstd = rsqrtf(vs * (1.f / NH) + EPSF);
#pragma unroll
  for (int u = 0; u < 4; ++u) {
    const int c = lane * 4 + u;
    float y = (x[u] - mean) * rstd * lnh_g[l * NH + c] + lnh_b[l * NH + c];
    h[row * NH + c] = hin[u] + fmaxf(y, 0.f);
  }
}

// ---------------------------------------------------------------------------
// e activation chain: t = silu(LN2(relu(LN1(e_new)))) -> bf16.
// ---------------------------------------------------------------------------
__global__ __launch_bounds__(256) void k_e_act(
    const float* __restrict__ enew, const float* __restrict__ lne_g,
    const float* __restrict__ lne_b, const float* __restrict__ plo_g,
    const float* __restrict__ plo_b, __bf16* __restrict__ tbf, int l) {
  const int warp = threadIdx.x >> 5, lane = threadIdx.x & 31;
  const long row = (long)blockIdx.x * 8 + warp;
  float x[4];
#pragma unroll
  for (int u = 0; u < 4; ++u) x[u] = enew[row * NH + lane * 4 + u];
  float s = warp_sum32(x[0] + x[1] + x[2] + x[3]);
  const float m1 = s * (1.f / NH);
  float vs = 0.f;
#pragma unroll
  for (int u = 0; u < 4; ++u) { float d = x[u] - m1; vs += d * d; }
  vs = warp_sum32(vs);
  const float r1 = rsqrtf(vs * (1.f / NH) + EPSF);
  float y[4];
#pragma unroll
  for (int u = 0; u < 4; ++u) {
    const int c = lane * 4 + u;
    float yv = (x[u] - m1) * r1 * lne_g[l * NH + c] + lne_b[l * NH + c];
    y[u] = fmaxf(yv, 0.f);
  }
  s = warp_sum32(y[0] + y[1] + y[2] + y[3]);
  const float m2 = s * (1.f / NH);
  vs = 0.f;
#pragma unroll
  for (int u = 0; u < 4; ++u) { float d = y[u] - m2; vs += d * d; }
  vs = warp_sum32(vs);
  const float r2 = rsqrtf(vs * (1.f / NH) + EPSF);
#pragma unroll
  for (int u = 0; u < 4; ++u) {
    const int c = lane * 4 + u;
    float z = (y[u] - m2) * r2 * plo_g[l * NH + c] + plo_b[l * NH + c];
    float tt = z / (1.f + __expf(-z));  // silu
    tbf[row * NH + c] = f2bf(tt);
  }
}

// ---------------------------------------------------------------------------
// plo GEMM + residual: e = e + t@plo_W + plo_bW  (writes f32 + bf16 mirror)
// ---------------------------------------------------------------------------
__global__ __launch_bounds__(128) void k_plo_gemm(
    const __bf16* __restrict__ tbf, const __bf16* __restrict__ wpt,
    const float* __restrict__ plobW, float* __restrict__ e,
    __bf16* __restrict__ ebf, int l) {
  __shared__ __align__(16) __bf16 sA[64][136];
  __shared__ __align__(16) __bf16 sBt[128][136];
  __shared__ float sb[NH];
  const long row0 = (long)blockIdx.x * 64;
  const int t = threadIdx.x;
  STAGE_TILE((uint32_t)(uintptr_t)&sA[0][0], tbf + row0 * NH, 64, t);
  STAGE_TILE((uint32_t)(uintptr_t)&sBt[0][0], wpt, 128, t);
  if (t < NH) sb[t] = plobW[l * NH + t];
  STAGE_WAIT(t);
  __syncthreads();

  v8f acc[8];
#pragma unroll
  for (int nt = 0; nt < 8; ++nt)
#pragma unroll
    for (int r = 0; r < 8; ++r) acc[nt][r] = 0.f;
  gemm_tile(sA, sBt, acc);

  const int w = t >> 5, lane = t & 31;
  const int half = lane >> 4, lq = lane & 15;
#pragma unroll
  for (int nt = 0; nt < 8; ++nt) {
    const int n = nt * 16 + lq;
#pragma unroll
    for (int r = 0; r < 8; ++r) {
      const int ml = 16 * w + r + 8 * half;
      const long row = row0 + ml;
      const float val = acc[nt][r] + sb[n] + e[row * NH + n];
      e[row * NH + n] = val;
      ebf[row * NH + n] = f2bf(val);
    }
  }
}

// ---------------------------------------------------------------------------
// Head: GroupNorm(32) stats / finalize / relu + 1x1 conv.
// ---------------------------------------------------------------------------
__global__ void k_zero(float* p, int n) {
  const int i = blockIdx.x * blockDim.x + threadIdx.x;
  if (i < n) p[i] = 0.f;
}

__global__ __launch_bounds__(256) void k_gn_stats(
    const float* __restrict__ e, float* __restrict__ sum,
    float* __restrict__ ssum) {
  __shared__ float ls[256], lss[256];
  const int blk = blockIdx.x;
  const int b = blk >> 9;  // 512 blocks per batch
  const long row0 = (long)blk * 128;
  const int t = threadIdx.x;
  const int c = t & 127, hh = t >> 7;
  float s = 0.f, ss = 0.f;
  for (int r = hh; r < 128; r += 2) {
    const float v = e[(row0 + r) * NH + c];
    s += v;
    ss += v * v;
  }
  ls[t] = s;
  lss[t] = ss;
  __syncthreads();
  if (t < 128) {
    atomicAdd(&sum[b * NH + c], ls[t] + ls[t + 128]);
    atomicAdd(&ssum[b * NH + c], lss[t] + lss[t + 128]);
  }
}

__global__ void k_gn_finalize(const float* __restrict__ sum,
                              const float* __restrict__ ssum,
                              float* __restrict__ mu,
                              float* __restrict__ rstd) {
  const int t = threadIdx.x;
  if (t < NB * NG) {
    const int b = t >> 5, g = t & 31;
    float s = 0.f, ss = 0.f;
#pragma unroll
    for (int u = 0; u < 4; ++u) {
      const int c = g * 4 + u;
      s += sum[b * NH + c];
      ss += ssum[b * NH + c];
    }
    const float cnt = 4.f * NVV;
    const float m = s / cnt;
    mu[t] = m;
    rstd[t] = rsqrtf(ss / cnt - m * m + EPSF);
  }
}

__global__ __launch_bounds__(256) void k_head_out(
    const float* __restrict__ e, const float* __restrict__ mu,
    const float* __restrict__ rstd, const float* __restrict__ gn_g,
    const float* __restrict__ gn_b, const float* __restrict__ W_out,
    const float* __restrict__ b_out, float* __restrict__ out) {
  const int warp = threadIdx.x >> 5, lane = threadIdx.x & 31;
  const long row = (long)blockIdx.x * 8 + warp;
  const int b = (int)(row / NVV);
  const float m = mu[b * NG + lane];  // c = 4*lane..4*lane+3 -> group = lane
  const float rs = rstd[b * NG + lane];
  float acc = 0.f;
#pragma unroll
  for (int u = 0; u < 4; ++u) {
    const int c = lane * 4 + u;
    float z = (e[row * NH + c] - m) * rs * gn_g[c] + gn_b[c];
    acc += fmaxf(z, 0.f) * W_out[c];
  }
  acc = warp_sum32(acc);
  if (lane == 0) out[row] = acc + b_out[0];
}

// ---------------------------------------------------------------------------
extern "C" void kernel_launch(void* const* d_in, const int* in_sizes, int n_in,
                              void* d_out, int out_size, void* d_ws,
                              size_t ws_size, hipStream_t stream) {
  (void)in_sizes; (void)n_in; (void)out_size; (void)ws_size;
  const int*   x       = (const int*)d_in[0];
  const float* adj     = (const float*)d_in[1];
  const float* emb     = (const float*)d_in[2];
  const float* W_node  = (const float*)d_in[3];
  const float* b_node  = (const float*)d_in[4];
  const float* W_eattr = (const float*)d_in[5];
  const float* b_eattr = (const float*)d_in[6];
  const float* W_ee    = (const float*)d_in[7];
  const float* b_ee    = (const float*)d_in[8];
  const float* Wu = (const float*)d_in[9];  const float* bu = (const float*)d_in[10];
  const float* Wv = (const float*)d_in[11]; const float* bv = (const float*)d_in[12];
  const float* Wa = (const float*)d_in[13]; const float* ba = (const float*)d_in[14];
  const float* Wb = (const float*)d_in[15]; const float* bb = (const float*)d_in[16];
  const float* Wc = (const float*)d_in[17]; const float* bc = (const float*)d_in[18];
  const float* lnh_g = (const float*)d_in[19]; const float* lnh_b = (const float*)d_in[20];
  const float* lne_g = (const float*)d_in[21]; const float* lne_b = (const float*)d_in[22];
  const float* plo_g = (const float*)d_in[23]; const float* plo_b = (const float*)d_in[24];
  const float* plo_W = (const float*)d_in[25]; const float* plo_bW = (const float*)d_in[26];
  const float* gn_g = (const float*)d_in[27]; const float* gn_b = (const float*)d_in[28];
  const float* W_out = (const float*)d_in[29]; const float* b_out = (const float*)d_in[30];
  float* out = (float*)d_out;

  char* ws = (char*)d_ws;
  size_t off = 0;
  auto alloc = [&](size_t bytes) -> void* {
    void* p = ws + off;
    off = (off + bytes + 255) & ~(size_t)255;
    return p;
  };
  const size_t NE = (size_t)NB * NVV * NH;  // 16.7M edge elements
  float*  e_f  = (float*)alloc(NE * 4);
  __bf16* e_bf = (__bf16*)alloc(NE * 2);
  float*  enew = (float*)alloc(NE * 4);
  __bf16* t_bf = (__bf16*)alloc(NE * 2);
  const size_t NHROWS = (size_t)NB * NV * NH;
  float* hbuf = (float*)alloc(NHROWS * 4);
  float* Uh   = (float*)alloc(NHROWS * 4);
  float* Vh   = (float*)alloc(NHROWS * 4);
  float* Ah   = (float*)alloc(NHROWS * 4);
  float* Bh   = (float*)alloc(NHROWS * 4);
  float* agg  = (float*)alloc(NHROWS * 4);
  __bf16* wbf = (__bf16*)alloc((size_t)7 * NH * NH * 2);  // transposed bf16 W
  float* gsum = (float*)alloc((size_t)NB * NH * 4);
  float* gss  = (float*)alloc((size_t)NB * NH * 4);
  float* gmu  = (float*)alloc((size_t)NB * NG * 4);
  float* grs  = (float*)alloc((size_t)NB * NG * 4);

  k_prep_w<<<7, 256, 0, stream>>>(W_ee, Wc, plo_W, wbf);
  k_node_embed<<<NB * NV, 128, 0, stream>>>(x, emb, W_node, b_node, hbuf);
  k_edge_embed<<<NB * NVV / 64, 128, 0, stream>>>(adj, W_eattr, b_eattr, wbf,
                                                  b_ee, e_f, e_bf);
  for (int l = 0; l < NL; ++l) {
    k_hmm<<<NB * NV, 128, 0, stream>>>(hbuf, Wu, bu, Wv, bv, Wa, ba, Wb, bb,
                                       Uh, Vh, Ah, Bh, agg, l);
    k_edge_gemm_gate<<<NB * NV * (NV / 64), 128, 0, stream>>>(
        e_bf, wbf + (size_t)(1 + l) * NH * NH, bc, Ah, Bh, Vh, adj, enew, agg,
        l);
    k_h_update<<<(NB * NV) / 8, 256, 0, stream>>>(Uh, agg, lnh_g, lnh_b, hbuf,
                                                  l);
    k_e_act<<<NB * NVV / 8, 256, 0, stream>>>(enew, lne_g, lne_b, plo_g, plo_b,
                                              t_bf, l);
    k_plo_gemm<<<NB * NVV / 64, 128, 0, stream>>>(
        t_bf, wbf + (size_t)(4 + l) * NH * NH, plo_bW, e_f, e_bf, l);
  }
  k_zero<<<1, 256, 0, stream>>>(gsum, NB * NH);
  k_zero<<<1, 256, 0, stream>>>(gss, NB * NH);
  k_gn_stats<<<NB * NVV / 128, 256, 0, stream>>>(e_f, gsum, gss);
  k_gn_finalize<<<1, 64, 0, stream>>>(gsum, gss, gmu, grs);
  k_head_out<<<NB * NVV / 8, 256, 0, stream>>>(e_f, gmu, grs, gn_g, gn_b,
                                               W_out, b_out, out);
}
